// HyperModulatedResidualBlock_42167988912487
// MI455X (gfx1250) — matrix operand order repaired and checked by reading, not compiled, not run on hardware
//
#include <hip/hip_runtime.h>

// ---------------------------------------------------------------------------
// Types
// ---------------------------------------------------------------------------
typedef __attribute__((ext_vector_type(16))) __bf16       v16bf;
typedef __attribute__((ext_vector_type(8)))  float        v8f;
typedef __attribute__((ext_vector_type(4)))  unsigned int u32x4;
typedef __attribute__((ext_vector_type(8)))  int          i32x8;
typedef __attribute__((ext_vector_type(4)))  int          i32x4;
typedef __attribute__((ext_vector_type(4)))  unsigned int u4;

union Frag16 { v16bf v; u4 q[2]; __bf16 e[16]; };

struct F2 { float mean; float rstd; };

#if defined(__has_builtin)
#  if __has_builtin(__builtin_amdgcn_tensor_load_to_lds)
#    define USE_TDM 1
#  else
#    define USE_TDM 0
#  endif
#else
#  define USE_TDM 0
#endif

#if !USE_TDM
#warning "TDM builtin __builtin_amdgcn_tensor_load_to_lds not available; using manual LDS staging"
#endif

__device__ __forceinline__ v8f wmma_bf16(v16bf a, v16bf b, v8f c) {
  return __builtin_amdgcn_wmma_f32_16x16x32_bf16(false, a, false, b, (short)0, c,
                                                 false, false);
}

#if USE_TDM
// 2D TDM load: tile of (tile_rows x tile_k_elems) bf16 elements, row stride
// row_stride_elems in global memory, written to LDS with a 16B pad after each
// 64B row (pad_interval=16 dwords, pad_amount=4 dwords) -> 80B LDS row stride.
__device__ __forceinline__ void tdm_load_tile_2d(unsigned lds_byte_off,
                                                 unsigned long long gaddr,
                                                 unsigned tile_k_elems,
                                                 unsigned tile_rows,
                                                 unsigned long long row_stride_elems) {
  u32x4 g0;
  g0[0] = 1u;                                              // count=1, user D#
  g0[1] = lds_byte_off;                                    // lds_addr
  g0[2] = (unsigned)(gaddr & 0xffffffffull);               // global_addr lo
  g0[3] = (unsigned)((gaddr >> 32) & 0x01ffffffull) | (2u << 30); // hi | type=2
  i32x8 g1;
  // data_size=1 (2B), pad_enable, pad_interval=3 (16 dwords), pad_amount=3 (4 dwords)
  g1[0] = (int)((1u << 16) | (1u << 20) | (3u << 22) | (3u << 25));
  const unsigned td0 = 0x7fffffffu;   // large tensor dims: tiles always in-bounds
  const unsigned td1 = 0x7fffffffu;
  g1[1] = (int)(td0 << 16);
  g1[2] = (int)((td0 >> 16) | (td1 << 16));
  g1[3] = (int)((td1 >> 16) | (tile_k_elems << 16));       // tile_dim0
  g1[4] = (int)(tile_rows & 0xffffu);                      // tile_dim1 (tile_dim2=0)
  g1[5] = (int)(unsigned)(row_stride_elems & 0xffffffffull);
  g1[6] = (int)(unsigned)((row_stride_elems >> 32) & 0xffffull);
  g1[7] = 0;
  i32x4 gz = {0, 0, 0, 0};
#if __clang_major__ >= 23
  i32x8 gz8 = {0, 0, 0, 0, 0, 0, 0, 0};
  __builtin_amdgcn_tensor_load_to_lds(g0, g1, gz, gz, gz8, 0);
#else
  __builtin_amdgcn_tensor_load_to_lds(g0, g1, gz, gz, 0);
#endif
}
#endif

// ---------------------------------------------------------------------------
// Kernel 1: f32 -> bf16 conversion (x)
// ---------------------------------------------------------------------------
__global__ __launch_bounds__(256) void f32_to_bf16_kernel(
    const float* __restrict__ in, __bf16* __restrict__ out, size_t n) {
  size_t i = (size_t)blockIdx.x * 256 + threadIdx.x;
  if (i < n) out[i] = (__bf16)in[i];
}

// ---------------------------------------------------------------------------
// Kernel 2: hyper-network head GEMM, fused bias+base add, bf16 weight output.
//   delta[b,n] = sum_k z[b,k]*head_w[n,k];  w[b,n] = base_w[n]+head_b[n]+delta
//   M=16 (batch) is exactly one WMMA tile; K=64 -> 2 x wmma_f32_16x16x32_bf16.
//   One wave per 16-wide N tile, 8 waves per block.
// ---------------------------------------------------------------------------
__global__ __launch_bounds__(256) void head_gemm_kernel(
    const float* __restrict__ z,       // [16, 64]
    const float* __restrict__ head_w,  // [N, 64]
    const float* __restrict__ head_b,  // [N]
    const float* __restrict__ base_w,  // [N]
    __bf16* __restrict__ wout,         // [16, N]
    int N) {
  const int wid  = threadIdx.x >> 5;
  const int lane = threadIdx.x & 31;
  const int n0   = (blockIdx.x * 8 + wid) * 16;
  if (n0 >= N) return;

  const int mrow = lane & 15;         // A-matrix: M = lane % 16
  const int koff = (lane >> 4) * 16;  // A: lanes>=16 hold K=16..31 (contiguous)

  // A fragments (z in bf16) for the two K=32 steps.
  Frag16 a[2];
#pragma unroll
  for (int s = 0; s < 2; ++s)
#pragma unroll
    for (int i = 0; i < 8; ++i) {
      a[s].e[i]     = (__bf16)z[mrow * 64 + s * 32 + koff + i];
      a[s].e[8 + i] = (__bf16)z[mrow * 64 + s * 32 + 8 + koff + i];
    }

  // B fragments: lane n = lane%16 -> row (n0+n) of head_w; lane half selects
  // K=0..15 vs K=16..31 (16 consecutive elements -> frag elems 0..15).
  const int nglob = n0 + (lane & 15);
  const int kh    = (lane >> 4) * 16;
  const float* wr = head_w + (size_t)nglob * 64;
  Frag16 b0, b1;
#pragma unroll
  for (int i = 0; i < 16; ++i) {
    b0.e[i] = (__bf16)wr[kh + i];        // K step 0:  k = kh + i
    b1.e[i] = (__bf16)wr[32 + kh + i];   // K step 1:  k = 32 + kh + i
  }

  v8f acc = {0, 0, 0, 0, 0, 0, 0, 0};
  acc = wmma_bf16(a[0].v, b0.v, acc);
  acc = wmma_bf16(a[1].v, b1.v, acc);

  const float addv = head_b[nglob] + base_w[nglob];
  const int rbase  = (lane >> 4) * 8;  // C/D: vgpr r -> M=r (+8 for lanes>=16)
#pragma unroll
  for (int r = 0; r < 8; ++r) {
    const int m = r + rbase;  // batch index
    wout[(size_t)m * N + nglob] = (__bf16)(acc[r] + addv);
  }
}

// ---------------------------------------------------------------------------
// Kernel 3: implicit-GEMM hyper conv (per-batch weights), bf16 in, f32 out.
//   Out [b, 256, 32, 32].  Block = 256 thr (8 waves) computes all 256 M rows
//   x 32 N columns, so the im2col gather is done exactly once per element.
//   Per wave: 2 M-subtiles x 2 N-subtiles (4 accumulators, 4 WMMAs / K-step).
//   Double-buffered LDS: A tile (256x32 bf16 weights) staged via the Tensor
//   Data Mover with LDS row padding (80B stride); B tile (im2col, stored
//   transposed [n][k]) staged by gathers. TDM + gathers for step k+1 overlap
//   the WMMAs of step k; one barrier per K-step.
// ---------------------------------------------------------------------------
template <int IN_C, int KSIZE, int STRIDE, int H_IN>
__global__ __launch_bounds__(256) void hyper_conv_wmma(
    const __bf16* __restrict__ xbf,  // [B, IN_C, H_IN, H_IN]
    const __bf16* __restrict__ wbf,  // [B, 256, IN_C*KSIZE*KSIZE]
    float* __restrict__ out) {       // [B, 256, 32, 32]
  constexpr int KTOT = IN_C * KSIZE * KSIZE;
  constexpr int PAD  = KSIZE / 2;
  constexpr int AST  = 40;  // LDS row stride in bf16 (80B) to dodge bank conflicts

  __shared__ __bf16 lA[2][256 * AST];
  __shared__ __bf16 lB[2][32 * AST];

  const int b    = blockIdx.z;
  const int n0   = blockIdx.x * 32;
  const int tid  = threadIdx.x;
  const int wid  = tid >> 5;
  const int lane = tid & 31;

  const __bf16* xb = xbf + (size_t)b * IN_C * H_IN * H_IN;
  const __bf16* wb = wbf + (size_t)b * 256 * KTOT;

  v8f acc[2][2];
#pragma unroll
  for (int im = 0; im < 2; ++im)
#pragma unroll
    for (int jn = 0; jn < 2; ++jn) acc[im][jn] = (v8f){0, 0, 0, 0, 0, 0, 0, 0};

  const int mrow  = lane & 15;
  const int koff  = (lane >> 4) * 8;
  const int bkoff = (lane >> 4) * 16;

  // ---- stagers -------------------------------------------------------------
  auto stage_A = [&](int buf, int kk) {
#if USE_TDM
    if (wid == 0) {
      tdm_load_tile_2d((unsigned)(size_t)&lA[buf][0],
                       (unsigned long long)(size_t)(wb + kk), 32u, 256u,
                       (unsigned long long)KTOT);
    }
#else
    const __bf16* src = wb + (size_t)tid * KTOT + kk;  // row = tid (256 rows)
    *(u4*)&lA[buf][tid * AST + 0]  = *(const u4*)(src);
    *(u4*)&lA[buf][tid * AST + 8]  = *(const u4*)(src + 8);
    *(u4*)&lA[buf][tid * AST + 16] = *(const u4*)(src + 16);
    *(u4*)&lA[buf][tid * AST + 24] = *(const u4*)(src + 24);
#endif
  };

  auto stage_B = [&](int buf, int kk) {
#pragma unroll
    for (int e = 0; e < 4; ++e) {
      const int idx = tid + e * 256;  // 32 n x 32 k = 1024 elements
      const int kl  = idx & 31;
      const int nl  = idx >> 5;
      const int k   = kk + kl;
      int c, khh, kww;
      if (KSIZE == 3) {
        c = k / 9;
        const int r = k - c * 9;
        khh = r / 3;
        kww = r - khh * 3;
      } else {
        c = k; khh = 0; kww = 0;
      }
      const int ng = n0 + nl;
      const int oh = ng >> 5;
      const int ow = ng & 31;
      const int ih = oh * STRIDE - PAD + khh;
      const int iw = ow * STRIDE - PAD + kww;
      __bf16 v = (__bf16)0.0f;
      if ((unsigned)ih < (unsigned)H_IN && (unsigned)iw < (unsigned)H_IN)
        v = xb[((size_t)c * H_IN + ih) * H_IN + iw];
      lB[buf][nl * AST + kl] = v;
    }
  };

  // ---- prologue: fill buffer 0 --------------------------------------------
  stage_B(0, 0);
  stage_A(0, 0);

  // ---- main pipeline -------------------------------------------------------
  for (int kk = 0; kk < KTOT; kk += 32) {
    const int cur = (kk >> 5) & 1;
    const int nxt = cur ^ 1;
#if USE_TDM
    if (wid == 0) __builtin_amdgcn_s_wait_tensorcnt(0);  // cur A-tile landed
#endif
    __syncthreads();  // cur visible to all; prior-step reads of nxt complete

    if (kk + 32 < KTOT) {  // prefetch next buffers, overlapped with WMMAs
      stage_B(nxt, kk + 32);
      stage_A(nxt, kk + 32);
    }

    Frag16 a0, a1;
    a0.q[0] = *(const u4*)&lA[cur][(wid * 16 + mrow) * AST + koff];
    a0.q[1] = *(const u4*)&lA[cur][(wid * 16 + mrow) * AST + 16 + koff];
    a1.q[0] = *(const u4*)&lA[cur][(128 + wid * 16 + mrow) * AST + koff];
    a1.q[1] = *(const u4*)&lA[cur][(128 + wid * 16 + mrow) * AST + 16 + koff];
#pragma unroll
    for (int jn = 0; jn < 2; ++jn) {
      Frag16 bf;
      bf.q[0] = *(const u4*)&lB[cur][(jn * 16 + mrow) * AST + koff];
      bf.q[1] = *(const u4*)&lB[cur][(jn * 16 + mrow) * AST + 16 + koff];
      acc[0][jn] = wmma_bf16(a0.v, bf.v, acc[0][jn]);
      acc[1][jn] = wmma_bf16(a1.v, bf.v, acc[1][jn]);
    }
  }

  // ---- store: C/D layout N=lane%16, M = vgpr + 8*(lane>=16) ----
  const int rbase = (lane >> 4) * 8;
  const int ncol  = lane & 15;
#pragma unroll
  for (int im = 0; im < 2; ++im) {
    float* ob = out + ((size_t)b * 256 + im * 128 + wid * 16) * 1024 + n0;
#pragma unroll
    for (int jn = 0; jn < 2; ++jn)
#pragma unroll
      for (int r = 0; r < 8; ++r)
        ob[(size_t)(r + rbase) * 1024 + jn * 16 + ncol] = acc[im][jn][r];
  }
}

// ---------------------------------------------------------------------------
// Kernel 4: GroupNorm statistics. One block per (b, group): 32 ch x 1024 px.
// ---------------------------------------------------------------------------
__global__ __launch_bounds__(256) void gn_stats_kernel(
    const float* __restrict__ xr, F2* __restrict__ stats) {
  const int bg = blockIdx.x;  // b*8 + g  (channels of a group are contiguous)
  const float* p = xr + (size_t)bg * 32 * 1024;
  float s = 0.f, s2 = 0.f;
  for (int i = threadIdx.x; i < 32768; i += 256) {
    const float v = p[i];
    s += v;
    s2 += v * v;
  }
  __shared__ float sh1[256], sh2[256];
  sh1[threadIdx.x] = s;
  sh2[threadIdx.x] = s2;
  __syncthreads();
  for (int off = 128; off > 0; off >>= 1) {
    if (threadIdx.x < (unsigned)off) {
      sh1[threadIdx.x] += sh1[threadIdx.x + off];
      sh2[threadIdx.x] += sh2[threadIdx.x + off];
    }
    __syncthreads();
  }
  if (threadIdx.x == 0) {
    const float mean = sh1[0] * (1.0f / 32768.0f);
    const float var  = sh2[0] * (1.0f / 32768.0f) - mean * mean;
    F2 st;
    st.mean = mean;
    st.rstd = rsqrtf(var + 1e-5f);
    stats[bg] = st;
  }
}

// ---------------------------------------------------------------------------
// Kernel 5: GN1 apply + ReLU + bf16 (input for conv2)
// ---------------------------------------------------------------------------
__global__ __launch_bounds__(256) void gn_relu_bf16_kernel(
    const float* __restrict__ xr, const F2* __restrict__ stats,
    const float* __restrict__ gw, const float* __restrict__ gb,
    __bf16* __restrict__ out) {
  const size_t idx = (size_t)blockIdx.x * 256 + threadIdx.x;
  const int c  = (int)((idx >> 10) & 255);
  const int bg = (int)(idx >> 15);
  const F2 st  = stats[bg];
  const float v = (xr[idx] - st.mean) * st.rstd * gw[c] + gb[c];
  out[idx] = (__bf16)fmaxf(v, 0.0f);
}

// ---------------------------------------------------------------------------
// Kernel 6: final = relu(gn2(conv2) + shortcut), f32 out
// ---------------------------------------------------------------------------
__global__ __launch_bounds__(256) void final_kernel(
    const float* __restrict__ c2, const F2* __restrict__ stats,
    const float* __restrict__ gw, const float* __restrict__ gb,
    const float* __restrict__ sc, float* __restrict__ out) {
  const size_t idx = (size_t)blockIdx.x * 256 + threadIdx.x;
  const int c  = (int)((idx >> 10) & 255);
  const int bg = (int)(idx >> 15);
  const F2 st  = stats[bg];
  const float v = (c2[idx] - st.mean) * st.rstd * gw[c] + gb[c] + sc[idx];
  out[idx] = fmaxf(v, 0.0f);
}

// ---------------------------------------------------------------------------
// Host launcher
// ---------------------------------------------------------------------------
extern "C" void kernel_launch(void* const* d_in, const int* in_sizes, int n_in,
                              void* d_out, int out_size, void* d_ws,
                              size_t ws_size, hipStream_t stream) {
  (void)in_sizes; (void)n_in; (void)out_size; (void)ws_size;

  const float* x        = (const float*)d_in[0];
  const float* z        = (const float*)d_in[1];
  const float* base_w1  = (const float*)d_in[2];
  const float* head_w1  = (const float*)d_in[3];
  const float* head_b1  = (const float*)d_in[4];
  const float* gn_w1    = (const float*)d_in[5];
  const float* gn_b1    = (const float*)d_in[6];
  const float* base_w2  = (const float*)d_in[7];
  const float* head_w2  = (const float*)d_in[8];
  const float* head_b2  = (const float*)d_in[9];
  const float* gn_w2    = (const float*)d_in[10];
  const float* gn_b2    = (const float*)d_in[11];
  const float* base_wsc = (const float*)d_in[12];
  const float* head_wsc = (const float*)d_in[13];
  const float* head_bsc = (const float*)d_in[14];
  float* out = (float*)d_out;

  constexpr int N1  = 256 * 128 * 9;  // 294912
  constexpr int N2  = 256 * 256 * 9;  // 589824
  constexpr int NSC = 256 * 128;      // 32768

  // Workspace layout (bytes)
  char* ws = (char*)d_ws;
  constexpr size_t OFF_W1  = 0;
  constexpr size_t SZ_W1   = (size_t)16 * N1 * 2;            //  9,437,184
  constexpr size_t OFF_W2  = OFF_W1 + SZ_W1;
  constexpr size_t SZ_W2   = (size_t)16 * N2 * 2;            // 18,874,368
  constexpr size_t OFF_WSC = OFF_W2 + SZ_W2;
  constexpr size_t SZ_WSC  = (size_t)16 * NSC * 2;           //  1,048,576
  constexpr size_t OFF_XB  = OFF_WSC + SZ_WSC;
  constexpr size_t SZ_XB   = (size_t)16 * 128 * 64 * 64 * 2; // 16,777,216
  constexpr size_t OFF_CR  = OFF_XB + SZ_XB;                 // conv raw f32 (reused)
  constexpr size_t SZ_CR   = (size_t)16 * 256 * 1024 * 4;    // 16,777,216
  constexpr size_t OFF_A1  = OFF_CR + SZ_CR;                 // act1 bf16
  constexpr size_t SZ_A1   = (size_t)16 * 256 * 1024 * 2;    //  8,388,608
  constexpr size_t OFF_SC  = OFF_A1 + SZ_A1;                 // shortcut raw f32
  constexpr size_t SZ_SC   = SZ_CR;
  constexpr size_t OFF_S1  = OFF_SC + SZ_SC;
  constexpr size_t OFF_S2  = OFF_S1 + 1024;

  __bf16* w1b   = (__bf16*)(ws + OFF_W1);
  __bf16* w2b   = (__bf16*)(ws + OFF_W2);
  __bf16* wscb  = (__bf16*)(ws + OFF_WSC);
  __bf16* xb    = (__bf16*)(ws + OFF_XB);
  float*  convr = (float*)(ws + OFF_CR);
  __bf16* a1b   = (__bf16*)(ws + OFF_A1);
  float*  scr   = (float*)(ws + OFF_SC);
  F2*     st1   = (F2*)(ws + OFF_S1);
  F2*     st2   = (F2*)(ws + OFF_S2);

  // 1) x -> bf16
  f32_to_bf16_kernel<<<32768, 256, 0, stream>>>(x, xb, (size_t)16 * 128 * 64 * 64);

  // 2) per-batch weights (WMMA GEMMs, fused base+bias add, bf16 out)
  head_gemm_kernel<<<N1 / 128, 256, 0, stream>>>(z, head_w1, head_b1, base_w1, w1b, N1);
  head_gemm_kernel<<<N2 / 128, 256, 0, stream>>>(z, head_w2, head_b2, base_w2, w2b, N2);
  head_gemm_kernel<<<NSC / 128, 256, 0, stream>>>(z, head_wsc, head_bsc, base_wsc, wscb, NSC);

  // 3) conv1 (3x3 s2) and shortcut (1x1 s2) implicit GEMMs
  hyper_conv_wmma<128, 3, 2, 64><<<dim3(32, 1, 16), 256, 0, stream>>>(xb, w1b, convr);
  hyper_conv_wmma<128, 1, 2, 64><<<dim3(32, 1, 16), 256, 0, stream>>>(xb, wscb, scr);

  // 4) GN1 + ReLU -> bf16 activations
  gn_stats_kernel<<<128, 256, 0, stream>>>(convr, st1);
  gn_relu_bf16_kernel<<<16384, 256, 0, stream>>>(convr, st1, gn_w1, gn_b1, a1b);

  // 5) conv2 (3x3 s1) — reuses convr buffer for its f32 output
  hyper_conv_wmma<256, 3, 1, 32><<<dim3(32, 1, 16), 256, 0, stream>>>(a1b, w2b, convr);

  // 6) GN2 + add shortcut + ReLU -> final f32 output
  gn_stats_kernel<<<128, 256, 0, stream>>>(convr, st2);
  final_kernel<<<16384, 256, 0, stream>>>(convr, st2, gn_w2, gn_b2, scr, out);
}
